// LigerLMHeadDPO_87299505259073
// MI455X (gfx1250) — compile-verified, matrix operand order checked by reading, not run on hardware
//
#include <hip/hip_runtime.h>
#include <math.h>

#define H_DIM 2048
#define V_DIM 32000
#define B_DIM 4
#define T_DIM 1024
#define BT    (B_DIM * T_DIM)      // 4096 tokens
#define NT    64                   // vocab rows per workgroup (LDS-resident)
#define NBLK  (V_DIM / NT)         // 500 vocab tiles
#define LDST  2056                 // LDS row stride in bf16 (2048 + 8 pad -> bank-conflict-free)
#define MT    256                  // token rows per M-chunk (8 waves x 32 rows)
#define IGNORE_INDEX (-100)
#define BETA  0.1f

typedef __attribute__((ext_vector_type(16))) __bf16 bf16x16;
typedef __attribute__((ext_vector_type(8)))  __bf16 bf16x8;
typedef __attribute__((ext_vector_type(8)))  float  floatx8;

// ---------------------------------------------------------------- helpers
__device__ inline unsigned bf_pack2(float a, float b) {
    unsigned ua = __builtin_bit_cast(unsigned, a);
    unsigned ub = __builtin_bit_cast(unsigned, b);
    ua = (ua + 0x7FFFu + ((ua >> 16) & 1u)) >> 16;   // RNE f32 -> bf16
    ub = (ub + 0x7FFFu + ((ub >> 16) & 1u)) >> 16;
    return (ub << 16) | (ua & 0xFFFFu);
}

__device__ inline bf16x16 frag16(const __bf16* lo, const __bf16* hi) {
    bf16x8 l = *(const bf16x8*)lo;
    bf16x8 h = *(const bf16x8*)hi;
    return __builtin_shufflevector(l, h, 0,1,2,3,4,5,6,7,8,9,10,11,12,13,14,15);
}

__device__ inline float wave_sum(float v) {
    for (int o = 1; o < 32; o <<= 1) v += __shfl_xor(v, o, 32);
    return v;
}

__device__ inline float block_reduce(float v, float* sh, bool domax) {
    int lane = threadIdx.x & 31, w = threadIdx.x >> 5;
    if (domax) { for (int o = 1; o < 32; o <<= 1) v = fmaxf(v, __shfl_xor(v, o, 32)); }
    else       { v = wave_sum(v); }
    if (lane == 0) sh[w] = v;
    __syncthreads();
    float r = domax ? -3.402823466e38f : 0.0f;
    int nw = (int)(blockDim.x >> 5);
    for (int i = 0; i < nw; ++i) r = domax ? fmaxf(r, sh[i]) : r + sh[i];
    __syncthreads();
    return r;
}

// ------------------------------------------------- kernel 1: f32 -> bf16
__global__ __launch_bounds__(256) void cvt_bf16_kernel(const float* __restrict__ in,
                                                       unsigned short* __restrict__ out,
                                                       int n4) {
    int i = blockIdx.x * blockDim.x + threadIdx.x;
    int stride = gridDim.x * blockDim.x;
    for (; i < n4; i += stride) {
        float4 f = ((const float4*)in)[i];
        uint2 p;
        p.x = bf_pack2(f.x, f.y);
        p.y = bf_pack2(f.z, f.w);
        ((uint2*)out)[i] = p;
    }
}

// --------------------------------- kernel 2: WMMA GEMM + softmax partials
// grid = (NBLK, 2); block = 256 threads (8 waves); dynamic LDS = NT*LDST*2 bytes
__global__ __launch_bounds__(256) void dpo_gemm_kernel(const __bf16* __restrict__ xb,
                                                       const __bf16* __restrict__ rxb,
                                                       const float*  __restrict__ W,
                                                       const float*  __restrict__ Wref,
                                                       float2* __restrict__ partials) {
    extern __shared__ char smem_raw[];
    __bf16* Wb = (__bf16*)smem_raw;

    const int mat = blockIdx.y;
    const __bf16* A  = mat ? rxb  : xb;
    const float*  Wg = mat ? Wref : W;
    const int n0  = blockIdx.x * NT;
    const int tid = threadIdx.x;

    // ---- stage the 64-row W slice into LDS as bf16 (read W exactly once) ----
    {
        int row = tid >> 2;           // 64 rows, 4 threads each
        int seg = tid & 3;            // 512 floats per thread
        const float* src = Wg + (size_t)(n0 + row) * H_DIM + seg * 512;
        __bf16* dst = Wb + row * LDST + seg * 512;
        #pragma unroll 4
        for (int j = 0; j < 512; j += 8) {
            float4 f0 = *(const float4*)(src + j);
            float4 f1 = *(const float4*)(src + j + 4);
            uint4 p;
            p.x = bf_pack2(f0.x, f0.y);
            p.y = bf_pack2(f0.z, f0.w);
            p.z = bf_pack2(f1.x, f1.y);
            p.w = bf_pack2(f1.z, f1.w);
            *(uint4*)(dst + j) = p;
        }
    }
    __syncthreads();

    const int wave = tid >> 5;
    const int lane = tid & 31;
    const int ln   = lane & 15;
    const int hi   = lane >> 4;

    // B fragment base pointers (constant across the whole kernel)
    const __bf16* bbase[4];
    #pragma unroll
    for (int nt = 0; nt < 4; ++nt)
        bbase[nt] = Wb + (nt * 16 + ln) * LDST + 8 * hi;

    for (int m0 = 0; m0 < BT; m0 += MT) {
        floatx8 c[2][4];
        #pragma unroll
        for (int mi = 0; mi < 2; ++mi)
            #pragma unroll
            for (int nt = 0; nt < 4; ++nt) {
                floatx8 z = {0.f,0.f,0.f,0.f,0.f,0.f,0.f,0.f};
                c[mi][nt] = z;
            }

        const int rowA = m0 + wave * 32 + ln;
        const __bf16* a0 = A + (size_t)rowA * H_DIM + 8 * hi;
        const __bf16* a1 = a0 + (size_t)16 * H_DIM;

        // ping-pong double buffer: set p computes while set 1-p loads.
        bf16x16 af[2][2], bfr[2][4];
        af[0][0] = frag16(a0, a0 + 16);
        af[0][1] = frag16(a1, a1 + 16);
        #pragma unroll
        for (int nt = 0; nt < 4; ++nt)
            bfr[0][nt] = frag16(bbase[nt], bbase[nt] + 16);

        for (int k = 0; k < H_DIM; k += 64) {
            __builtin_prefetch(a0 + k + 256, 0, 3);   // global_prefetch_b8 (GL2)
            __builtin_prefetch(a1 + k + 256, 0, 3);

            // --- load set 1 @ k+32, compute on set 0 @ k ---
            af[1][0] = frag16(a0 + k + 32, a0 + k + 48);
            af[1][1] = frag16(a1 + k + 32, a1 + k + 48);
            #pragma unroll
            for (int nt = 0; nt < 4; ++nt)
                bfr[1][nt] = frag16(bbase[nt] + k + 32, bbase[nt] + k + 48);

            #pragma unroll
            for (int mi = 0; mi < 2; ++mi)
                #pragma unroll
                for (int nt = 0; nt < 4; ++nt)
                    c[mi][nt] = __builtin_amdgcn_wmma_f32_16x16x32_bf16(
                        false, af[0][mi], false, bfr[0][nt], (short)0, c[mi][nt], false, false);

            // --- load set 0 @ k+64 (one-step overread at the tail is benign),
            //     compute on set 1 @ k+32 ---
            af[0][0] = frag16(a0 + k + 64, a0 + k + 80);
            af[0][1] = frag16(a1 + k + 64, a1 + k + 80);
            #pragma unroll
            for (int nt = 0; nt < 4; ++nt)
                bfr[0][nt] = frag16(bbase[nt] + k + 64, bbase[nt] + k + 80);

            #pragma unroll
            for (int mi = 0; mi < 2; ++mi)
                #pragma unroll
                for (int nt = 0; nt < 4; ++nt)
                    c[mi][nt] = __builtin_amdgcn_wmma_f32_16x16x32_bf16(
                        false, af[1][mi], false, bfr[1][nt], (short)0, c[mi][nt], false, false);
        }

        // ---- per-row (max, sum-exp) over this 64-wide vocab slice ----
        // C layout: VGPR r holds row r (lanes 0-15) / row r+8 (lanes 16-31), N = lane&15
        #pragma unroll
        for (int mi = 0; mi < 2; ++mi) {
            float rmax[8], rsum[8];
            #pragma unroll
            for (int r = 0; r < 8; ++r) {
                float v = fmaxf(fmaxf(c[mi][0][r], c[mi][1][r]),
                                fmaxf(c[mi][2][r], c[mi][3][r]));
                for (int o = 1; o < 16; o <<= 1) v = fmaxf(v, __shfl_xor(v, o, 32));
                float s = 0.f;
                #pragma unroll
                for (int nt = 0; nt < 4; ++nt) s += __expf(c[mi][nt][r] - v);
                for (int o = 1; o < 16; o <<= 1) s += __shfl_xor(s, o, 32);
                rmax[r] = v; rsum[r] = s;
            }
            if (ln == 0) {
                int rowbase = m0 + wave * 32 + mi * 16 + 8 * hi;
                float2* pp = partials + ((size_t)(mat * BT + rowbase)) * NBLK + blockIdx.x;
                #pragma unroll
                for (int r = 0; r < 8; ++r)
                    pp[(size_t)r * NBLK] = make_float2(rmax[r], rsum[r]);
            }
        }
    }
}

// --------------- kernel 3: merge partials + target logit -> per-token logp
// grid = (BT, 2); block = 256
__global__ __launch_bounds__(256) void dpo_merge_kernel(const float2* __restrict__ partials,
                                                        const float* __restrict__ x,
                                                        const float* __restrict__ rx,
                                                        const float* __restrict__ W,
                                                        const float* __restrict__ Wref,
                                                        const int*  __restrict__ y,
                                                        float* __restrict__ tok_logp) {
    __shared__ float sh[8];
    const int t   = blockIdx.x;
    const int mat = blockIdx.y;
    const float* xg = mat ? rx   : x;
    const float* Wg = mat ? Wref : W;
    const float2* pp = partials + (size_t)(mat * BT + t) * NBLK;
    const int tid = threadIdx.x;

    float m = -3.402823466e38f;
    for (int i = tid; i < NBLK; i += 256) m = fmaxf(m, pp[i].x);
    m = block_reduce(m, sh, true);

    float s = 0.f;
    for (int i = tid; i < NBLK; i += 256) {
        float2 v = pp[i];
        s += v.y * __expf(v.x - m);
    }
    s = block_reduce(s, sh, false);

    const int yv = y[t];
    const bool valid = (yv != IGNORE_INDEX);
    float d = 0.f;
    {
        int yi = valid ? yv : 0;
        const float* xr = xg + (size_t)t  * H_DIM + tid * 8;
        const float* wr = Wg + (size_t)yi * H_DIM + tid * 8;
        float4 a0 = *(const float4*)(xr);
        float4 a1 = *(const float4*)(xr + 4);
        float4 b0 = *(const float4*)(wr);
        float4 b1 = *(const float4*)(wr + 4);
        d = a0.x*b0.x + a0.y*b0.y + a0.z*b0.z + a0.w*b0.w
          + a1.x*b1.x + a1.y*b1.y + a1.z*b1.z + a1.w*b1.w;
    }
    d = block_reduce(d, sh, false);

    if (tid == 0)
        tok_logp[mat * BT + t] = valid ? (d - (m + __logf(s))) : 0.f;
}

// ----------------------- kernel 4: sequence averages + DPO loss (1 block)
__global__ __launch_bounds__(256) void dpo_final_kernel(const float* __restrict__ tok_logp,
                                                        const int* __restrict__ y,
                                                        float* __restrict__ out) {
    __shared__ float sh[8];
    __shared__ float seq_lp[8];
    for (int s = 0; s < 8; ++s) {        // s = mat*4 + b
        int mat = s >> 2, b = s & 3;
        float acc = 0.f, cnt = 0.f;
        for (int i = threadIdx.x; i < T_DIM; i += 256) {
            int t = b * T_DIM + i;
            acc += tok_logp[mat * BT + t];
            cnt += (y[t] != IGNORE_INDEX) ? 1.f : 0.f;
        }
        acc = block_reduce(acc, sh, false);
        cnt = block_reduce(cnt, sh, false);
        if (threadIdx.x == 0) seq_lp[s] = acc / fmaxf(cnt, 1.f);
        __syncthreads();
    }
    if (threadIdx.x == 0) {
        float loss = 0.f;
        for (int i = 0; i < 2; ++i) {    // half = B/2
            float dlt = BETA * ((seq_lp[0 + i] - seq_lp[4 + i])
                              - (seq_lp[2 + i] - seq_lp[6 + i]));
            float ls = fminf(dlt, 0.f) - log1pf(__expf(-fabsf(dlt)));  // log_sigmoid
            loss += -ls;
        }
        out[0] = loss * 0.5f;
    }
}

// -------------------------------------------------------------- launcher
extern "C" void kernel_launch(void* const* d_in, const int* in_sizes, int n_in,
                              void* d_out, int out_size, void* d_ws, size_t ws_size,
                              hipStream_t stream) {
    const float* x    = (const float*)d_in[0];
    const float* rx   = (const float*)d_in[1];
    const int*   y    = (const int*)d_in[2];
    const float* W    = (const float*)d_in[3];
    const float* Wref = (const float*)d_in[4];
    float* out = (float*)d_out;

    char* w = (char*)d_ws;
    __bf16* xb  = (__bf16*)w;  w += (size_t)BT * H_DIM * 2;
    __bf16* rxb = (__bf16*)w;  w += (size_t)BT * H_DIM * 2;
    float2* partials = (float2*)w;  w += (size_t)2 * BT * NBLK * sizeof(float2);
    float* tok_logp = (float*)w;

    const int n4 = BT * H_DIM / 4;
    cvt_bf16_kernel<<<2048, 256, 0, stream>>>(x,  (unsigned short*)xb,  n4);
    cvt_bf16_kernel<<<2048, 256, 0, stream>>>(rx, (unsigned short*)rxb, n4);

    const size_t lds_bytes = (size_t)NT * LDST * sizeof(__bf16);   // 263,168 B of 320 KB
    dpo_gemm_kernel<<<dim3(NBLK, 2), 256, lds_bytes, stream>>>(xb, rxb, W, Wref, partials);

    dpo_merge_kernel<<<dim3(BT, 2), 256, 0, stream>>>(partials, x, rx, W, Wref, y, tok_logp);

    dpo_final_kernel<<<1, 256, 0, stream>>>(tok_logp, y, out);
}
